// TransformerBlock_49709951484076
// MI455X (gfx1250) — matrix operand order repaired and checked by reading, not compiled, not run on hardware
//
#include <hip/hip_runtime.h>
#include <hip/hip_bf16.h>

typedef __attribute__((ext_vector_type(16))) __bf16 v16bf;
typedef __attribute__((ext_vector_type(8)))  __bf16 v8bf;
typedef __attribute__((ext_vector_type(8)))  float  v8f;
typedef int v4i_vec __attribute__((vector_size(16)));

#define N_HEADS 16
#define EMBED   1024
#define SEQ     2048
#define BATCH   4
#define DHEAD   64
#define NTOK    (BATCH * SEQ)      // 8192
#define FFN     (4 * EMBED)        // 4096

// ---------------------------------------------------------------------------
// CDNA5 async global->LDS (ASYNCcnt-tracked, no VGPR staging), with fallback.
// Builtin signature (probe round 2): (v4i addrspace(1)*, v4i addrspace(3)*,
// imm offset, imm cpol).
// ---------------------------------------------------------------------------
#if defined(__has_builtin)
#if __has_builtin(__builtin_amdgcn_global_load_async_to_lds_b128)
#define HAVE_ASYNC_LDS 1
#endif
#if __has_builtin(__builtin_amdgcn_s_wait_asynccnt)
#define HAVE_WAIT_ASYNC 1
#endif
#endif

__device__ __forceinline__ void async_copy16(const void* g, void* l) {
#ifdef HAVE_ASYNC_LDS
  __builtin_amdgcn_global_load_async_to_lds_b128(
      (__attribute__((address_space(1))) v4i_vec*)(void*)g,
      (__attribute__((address_space(3))) v4i_vec*)l, 0, 0);
#else
  *(v4i_vec*)l = *(const v4i_vec*)g;
#endif
}

__device__ __forceinline__ void wait_async0() {
#ifdef HAVE_ASYNC_LDS
#ifdef HAVE_WAIT_ASYNC
  __builtin_amdgcn_s_wait_asynccnt(0);
#else
  asm volatile("s_wait_asynccnt 0" ::: "memory");
#endif
#endif
}

// ---------------------------------------------------------------------------
// Constant-mask lane XOR via ds_swizzle (group-of-32: imm = xor<<10 | and=0x1f)
// ---------------------------------------------------------------------------
template <int MASK>
__device__ __forceinline__ float swz_xor(float x) {
  return __int_as_float(
      __builtin_amdgcn_ds_swizzle(__float_as_int(x), (MASK << 10) | 0x1f));
}
__device__ __forceinline__ float rowmax16(float m) {
  m = fmaxf(m, swz_xor<1>(m));
  m = fmaxf(m, swz_xor<2>(m));
  m = fmaxf(m, swz_xor<4>(m));
  m = fmaxf(m, swz_xor<8>(m));
  return m;
}
__device__ __forceinline__ float rowsum16(float s) {
  s += swz_xor<1>(s);
  s += swz_xor<2>(s);
  s += swz_xor<4>(s);
  s += swz_xor<8>(s);
  return s;
}

// ---------------------------------------------------------------------------
// Fragment builder: two 16B reads -> one v16bf WMMA operand (ISA 7.12.2).
// ---------------------------------------------------------------------------
__device__ __forceinline__ v16bf mk_frag(const __bf16* p0, const __bf16* p1) {
  v8bf lo = *(const v8bf*)p0;
  v8bf hi = *(const v8bf*)p1;
  v16bf f;
#pragma unroll
  for (int i = 0; i < 8; ++i) { f[i] = lo[i]; f[i + 8] = hi[i]; }
  return f;
}

__device__ __forceinline__ v8f wmma_bf16(v16bf a, v16bf b, v8f c) {
  return __builtin_amdgcn_wmma_f32_16x16x32_bf16(false, a, false, b, (short)0, c,
                                                 false, false);
}

__device__ __forceinline__ float gelu_tanh(float x) {
  return 0.5f * x * (1.0f + tanhf(0.7978845608028654f * (x + 0.044715f * x * x * x)));
}

// ---------------------------------------------------------------------------
// fp32 -> bf16 convert (weights)
// ---------------------------------------------------------------------------
__global__ __launch_bounds__(256) void cvt_bf16(const float* __restrict__ in,
                                                __bf16* __restrict__ out, int n4) {
  int i = blockIdx.x * 256 + threadIdx.x;
  if (i < n4) {
    float4 v = *(const float4*)&in[(size_t)i * 4];
    __bf16* o = out + (size_t)i * 4;
    o[0] = (__bf16)v.x; o[1] = (__bf16)v.y; o[2] = (__bf16)v.z; o[3] = (__bf16)v.w;
  }
}

// ---------------------------------------------------------------------------
// LayerNorm over d=1024, one row per block (256 thr = 8 waves), bf16 out.
// ---------------------------------------------------------------------------
__global__ __launch_bounds__(256) void ln_bf16(const float* __restrict__ X,
                                               const float* __restrict__ sc,
                                               const float* __restrict__ sh,
                                               __bf16* __restrict__ out) {
  __shared__ float red[16];
  int row = blockIdx.x, t = threadIdx.x;
  const float* xr = X + (size_t)row * EMBED;
  float4 x = *(const float4*)&xr[t * 4];
  float s  = x.x + x.y + x.z + x.w;
  float s2 = x.x * x.x + x.y * x.y + x.z * x.z + x.w * x.w;
#pragma unroll
  for (int o = 1; o < 32; o <<= 1) {
    s  += __shfl_xor(s,  o, 32);
    s2 += __shfl_xor(s2, o, 32);
  }
  int wave = t >> 5;
  if ((t & 31) == 0) { red[wave] = s; red[8 + wave] = s2; }
  __syncthreads();
  float ts = 0.f, ts2 = 0.f;
#pragma unroll
  for (int i = 0; i < 8; ++i) { ts += red[i]; ts2 += red[8 + i]; }
  float mean = ts * (1.0f / EMBED);
  float var  = ts2 * (1.0f / EMBED) - mean * mean;
  float rstd = rsqrtf(var + 1e-5f);
  float4 scv = *(const float4*)&sc[t * 4];
  float4 shv = *(const float4*)&sh[t * 4];
  __bf16* o = out + (size_t)row * EMBED + t * 4;
  o[0] = (__bf16)((x.x - mean) * rstd * scv.x + shv.x);
  o[1] = (__bf16)((x.y - mean) * rstd * scv.y + shv.y);
  o[2] = (__bf16)((x.z - mean) * rstd * scv.z + shv.z);
  o[3] = (__bf16)((x.w - mean) * rstd * scv.w + shv.w);
}

// ---------------------------------------------------------------------------
// bf16 WMMA GEMM: C = A[MxK] @ B[KxN], tile 128x128, BK=32.
// 8 waves as 2(m) x 4(n); each wave 64x32 = 4x2 WMMA tiles (8 acc).
// A panel staged with GLOBAL_LOAD_ASYNC_TO_LDS_B128; B transposed manually.
// Epilogues: 0 QKV head-scatter bf16 | 1 fp32 resid+acc
//            2 bf16 gelu(acc+bias)   | 3 fp32 resid+acc+bias
// ---------------------------------------------------------------------------
#define BM 128
#define BN 128
#define BK 32

template <int EPI>
__global__ __launch_bounds__(256) void gemm_bf16(
    const __bf16* __restrict__ A, const __bf16* __restrict__ B, int M, int N, int K,
    const float* __restrict__ resid, const float* __restrict__ bias,
    float* __restrict__ outf, __bf16* __restrict__ outb) {
  __shared__ __bf16 As[BM * BK];     // row-major [m][k]
  __shared__ __bf16 Bs[BN * BK];     // transposed [n][k]
  int t = threadIdx.x;
  int wave = t >> 5, lane = t & 31, half = lane >> 4, l16 = lane & 15;
  int wm = wave >> 2, wn = wave & 3;                  // 2 x 4 waves
  int m0 = blockIdx.y * BM, n0 = blockIdx.x * BN;

  v8f zero = {0.f, 0.f, 0.f, 0.f, 0.f, 0.f, 0.f, 0.f};
  v8f acc[4][2];
#pragma unroll
  for (int i = 0; i < 4; ++i)
#pragma unroll
    for (int j = 0; j < 2; ++j) acc[i][j] = zero;

  for (int k0 = 0; k0 < K; k0 += BK) {
    // A: 128x32 = 4096 bf16, async 16B per thread per pass, 2 passes
#pragma unroll
    for (int i = 0; i < 2; ++i) {
      int idx = i * 2048 + t * 8;
      int r = idx >> 5, c = idx & 31;
      async_copy16(&A[(size_t)(m0 + r) * K + k0 + c], &As[r * BK + c]);
    }
    // B: 32x128 = 4096 bf16, manual transpose into [n][k], 2 passes
#pragma unroll
    for (int i = 0; i < 2; ++i) {
      int idx = i * 2048 + t * 8;
      int kk = idx >> 7;             // /128
      int nn = idx & 127;
      v8bf bv = *(const v8bf*)&B[(size_t)(k0 + kk) * N + n0 + nn];
#pragma unroll
      for (int j = 0; j < 8; ++j) Bs[(nn + j) * BK + kk] = bv[j];
    }
    if (k0 + BK < K)                 // prefetch next A panel (global_prefetch_b8)
      __builtin_prefetch(&A[(size_t)(m0 + (t >> 1)) * K + k0 + BK], 0, 1);
    wait_async0();
    __syncthreads();

    v16bf afrag[4], bfrag[2];
#pragma unroll
    for (int i = 0; i < 4; ++i) {
      const __bf16* rp = &As[(wm * 64 + i * 16 + l16) * BK];
      afrag[i] = mk_frag(rp + half * 8, rp + 16 + half * 8);
    }
#pragma unroll
    for (int j = 0; j < 2; ++j) {
      const __bf16* rp = &Bs[(wn * 32 + j * 16 + l16) * BK];
      bfrag[j] = mk_frag(rp + half * 8, rp + 16 + half * 8);
    }
#pragma unroll
    for (int i = 0; i < 4; ++i)
#pragma unroll
      for (int j = 0; j < 2; ++j)
        acc[i][j] = wmma_bf16(afrag[i], bfrag[j], acc[i][j]);
    __syncthreads();
  }

  // C layout: VGPR r, lanes 0-15 -> M=r, lanes 16-31 -> M=r+8; N = lane%16.
#pragma unroll
  for (int i = 0; i < 4; ++i)
#pragma unroll
    for (int j = 0; j < 2; ++j)
#pragma unroll
      for (int r = 0; r < 8; ++r) {
        int gm = m0 + wm * 64 + i * 16 + r + half * 8;
        int gn = n0 + wn * 32 + j * 16 + l16;
        float v = acc[i][j][r];
        if constexpr (EPI == 0) {
          int b  = gm >> 11;          // /SEQ
          int sr = gm & (SEQ - 1);
          int h  = gn >> 6;           // /DHEAD
          int dh = gn & (DHEAD - 1);
          outb[((size_t)(b * N_HEADS + h) * SEQ + sr) * DHEAD + dh] = (__bf16)v;
        } else if constexpr (EPI == 1) {
          size_t o = (size_t)gm * N + gn;
          outf[o] = resid[o] + v;
        } else if constexpr (EPI == 2) {
          outb[(size_t)gm * N + gn] = (__bf16)gelu_tanh(v + bias[gn]);
        } else {
          size_t o = (size_t)gm * N + gn;
          outf[o] = resid[o] + v + bias[gn];
        }
      }
}

// ---------------------------------------------------------------------------
// Flash attention (causal): grid (SEQ/128, H, B), 256 thr = 8 waves.
// Wave w owns 16 query rows; group streams shared 32-key K/V tiles.
// K tile staged async; softmax stats in C layout via ds_swizzle reductions;
// P re-laid out via per-wave LDS; ctx += P@V.
// ---------------------------------------------------------------------------
__global__ __launch_bounds__(256) void flash_attn(const __bf16* __restrict__ Qg,
                                                  const __bf16* __restrict__ Kg,
                                                  const __bf16* __restrict__ Vg,
                                                  __bf16* __restrict__ CTX) {
  __shared__ __bf16 Ks[32 * DHEAD];      // [key][dh]   row-major
  __shared__ __bf16 Vs[DHEAD * 32];      // [dh][key]   transposed
  __shared__ __bf16 Ps[8][16 * 32];      // per-wave P  [qrow][key]

  int t = threadIdx.x;
  int wave = t >> 5, lane = t & 31, half = lane >> 4, l16 = lane & 15;
  int b = blockIdx.z, h = blockIdx.y;
  size_t bh = (size_t)(b * N_HEADS + h) * SEQ * DHEAD;
  int q0  = blockIdx.x * 128;
  int q0w = q0 + wave * 16;

  const __bf16* qrow = Qg + bh + (size_t)(q0w + l16) * DHEAD;
  v16bf qf[2];
#pragma unroll
  for (int kk = 0; kk < 2; ++kk)
    qf[kk] = mk_frag(qrow + kk * 32 + half * 8, qrow + kk * 32 + 16 + half * 8);

  v8f zero = {0.f, 0.f, 0.f, 0.f, 0.f, 0.f, 0.f, 0.f};
  v8f ctx[4] = {zero, zero, zero, zero};
  float mrow[8], lrow[8];
#pragma unroll
  for (int r = 0; r < 8; ++r) { mrow[r] = -1e30f; lrow[r] = 0.f; }

  int nkt = (q0 + 128) >> 5;
  for (int kt = 0; kt < nkt; ++kt) {
    int kb = kt * 32;
    {
      int key = t >> 3;
      int d0  = (t & 7) * 8;
      async_copy16(&Kg[bh + (size_t)(kb + key) * DHEAD + d0], &Ks[key * DHEAD + d0]);
      v8bf vv = *(const v8bf*)&Vg[bh + (size_t)(kb + key) * DHEAD + d0];
#pragma unroll
      for (int j = 0; j < 8; ++j) Vs[(d0 + j) * 32 + key] = vv[j];
    }
    wait_async0();
    __syncthreads();

    if (kb <= q0w + 15) {              // wave-uniform causal bound
      v8f s0 = zero, s1 = zero;
#pragma unroll
      for (int kk = 0; kk < 2; ++kk) {
        const __bf16* r0 = &Ks[l16 * DHEAD + kk * 32];
        const __bf16* r1 = &Ks[(16 + l16) * DHEAD + kk * 32];
        s0 = wmma_bf16(qf[kk], mk_frag(r0 + half * 8, r0 + 16 + half * 8), s0);
        s1 = wmma_bf16(qf[kk], mk_frag(r1 + half * 8, r1 + 16 + half * 8), s1);
      }

      float p0[8], p1[8];
#pragma unroll
      for (int r = 0; r < 8; ++r) {
        int q = q0w + r + half * 8;
        float a = s0[r] * 0.125f;      // 1/sqrt(64)
        float c = s1[r] * 0.125f;
        if (kb + l16 > q)      a = -1e30f;
        if (kb + 16 + l16 > q) c = -1e30f;
        p0[r] = a; p1[r] = c;
      }
#pragma unroll
      for (int r = 0; r < 8; ++r) {
        float m  = rowmax16(fmaxf(p0[r], p1[r]));
        float mn = fmaxf(mrow[r], m);
        float alpha = __expf(mrow[r] - mn);
        mrow[r] = mn;
        float e0 = __expf(p0[r] - mn);
        float e1 = __expf(p1[r] - mn);
        p0[r] = e0; p1[r] = e1;
        float ls = rowsum16(e0 + e1);
        lrow[r] = lrow[r] * alpha + ls;
#pragma unroll
        for (int n = 0; n < 4; ++n) ctx[n][r] *= alpha;
      }

#pragma unroll
      for (int r = 0; r < 8; ++r) {
        int pr = (r + 8 * half) * 32;
        Ps[wave][pr + l16]      = (__bf16)p0[r];
        Ps[wave][pr + 16 + l16] = (__bf16)p1[r];
      }
      asm volatile("s_wait_dscnt 0" ::: "memory");

      const __bf16* prw = &Ps[wave][l16 * 32];
      v16bf pf = mk_frag(prw + half * 8, prw + 16 + half * 8);
#pragma unroll
      for (int n = 0; n < 4; ++n) {
        const __bf16* vr = &Vs[(n * 16 + l16) * 32];
        ctx[n] = wmma_bf16(pf, mk_frag(vr + half * 8, vr + 16 + half * 8), ctx[n]);
      }
    }
    __syncthreads();
  }

  float rinv[8];
#pragma unroll
  for (int r = 0; r < 8; ++r) rinv[r] = 1.0f / lrow[r];
#pragma unroll
  for (int n = 0; n < 4; ++n)
#pragma unroll
    for (int r = 0; r < 8; ++r) {
      int q = q0w + r + half * 8;
      CTX[((size_t)b * SEQ + q) * EMBED + h * DHEAD + n * 16 + l16] =
          (__bf16)(ctx[n][r] * rinv[r]);
    }
}

// ---------------------------------------------------------------------------
// Orchestration
// ---------------------------------------------------------------------------
extern "C" void kernel_launch(void* const* d_in, const int* in_sizes, int n_in,
                              void* d_out, int out_size, void* d_ws, size_t ws_size,
                              hipStream_t stream) {
  const float* X    = (const float*)d_in[0];
  const float* Wq   = (const float*)d_in[1];
  const float* Wk   = (const float*)d_in[2];
  const float* Wv   = (const float*)d_in[3];
  const float* Wo   = (const float*)d_in[4];
  const float* W1   = (const float*)d_in[5];
  const float* b1   = (const float*)d_in[6];
  const float* W2   = (const float*)d_in[7];
  const float* b2   = (const float*)d_in[8];
  const float* ln1s = (const float*)d_in[9];
  const float* ln1b = (const float*)d_in[10];
  const float* ln2s = (const float*)d_in[11];
  const float* ln2b = (const float*)d_in[12];

  char* p = (char*)d_ws;
  auto carve = [&](size_t bytes) {
    void* r = (void*)p;
    p += (bytes + 255) & ~(size_t)255;
    return r;
  };
  __bf16* Xln  = (__bf16*)carve((size_t)NTOK * EMBED * 2);
  __bf16* Wqb  = (__bf16*)carve((size_t)EMBED * EMBED * 2);
  __bf16* Wkb  = (__bf16*)carve((size_t)EMBED * EMBED * 2);
  __bf16* Wvb  = (__bf16*)carve((size_t)EMBED * EMBED * 2);
  __bf16* Wob  = (__bf16*)carve((size_t)EMBED * EMBED * 2);
  __bf16* W1b  = (__bf16*)carve((size_t)EMBED * FFN * 2);
  __bf16* W2b  = (__bf16*)carve((size_t)FFN * EMBED * 2);
  __bf16* Qb   = (__bf16*)carve((size_t)NTOK * EMBED * 2);
  __bf16* Kb   = (__bf16*)carve((size_t)NTOK * EMBED * 2);
  __bf16* Vb   = (__bf16*)carve((size_t)NTOK * EMBED * 2);
  __bf16* Ctx  = (__bf16*)carve((size_t)NTOK * EMBED * 2);
  float*  X1   = (float*) carve((size_t)NTOK * EMBED * 4);
  __bf16* X1ln = (__bf16*)carve((size_t)NTOK * EMBED * 2);
  __bf16* Hb   = (__bf16*)carve((size_t)NTOK * FFN * 2);

  auto cvt = [&](const float* src, __bf16* dst, int n) {
    int n4 = n / 4;
    cvt_bf16<<<(n4 + 255) / 256, 256, 0, stream>>>(src, dst, n4);
  };
  cvt(Wq, Wqb, EMBED * EMBED);
  cvt(Wk, Wkb, EMBED * EMBED);
  cvt(Wv, Wvb, EMBED * EMBED);
  cvt(Wo, Wob, EMBED * EMBED);
  cvt(W1, W1b, EMBED * FFN);
  cvt(W2, W2b, FFN * EMBED);

  ln_bf16<<<NTOK, 256, 0, stream>>>(X, ln1s, ln1b, Xln);

  dim3 gProj(EMBED / BN, NTOK / BM);
  gemm_bf16<0><<<gProj, 256, 0, stream>>>(Xln, Wqb, NTOK, EMBED, EMBED,
                                          nullptr, nullptr, nullptr, Qb);
  gemm_bf16<0><<<gProj, 256, 0, stream>>>(Xln, Wkb, NTOK, EMBED, EMBED,
                                          nullptr, nullptr, nullptr, Kb);
  gemm_bf16<0><<<gProj, 256, 0, stream>>>(Xln, Wvb, NTOK, EMBED, EMBED,
                                          nullptr, nullptr, nullptr, Vb);

  flash_attn<<<dim3(SEQ / 128, N_HEADS, BATCH), 256, 0, stream>>>(Qb, Kb, Vb, Ctx);

  gemm_bf16<1><<<gProj, 256, 0, stream>>>(Ctx, Wob, NTOK, EMBED, EMBED,
                                          X, nullptr, X1, nullptr);

  ln_bf16<<<NTOK, 256, 0, stream>>>(X1, ln2s, ln2b, X1ln);

  gemm_bf16<2><<<dim3(FFN / BN, NTOK / BM), 256, 0, stream>>>(
      X1ln, W1b, NTOK, FFN, EMBED, nullptr, b1, nullptr, Hb);

  gemm_bf16<3><<<gProj, 256, 0, stream>>>(Hb, W2b, NTOK, EMBED, FFN,
                                          X1, b2, (float*)d_out, nullptr);
}
